// FlexRoPEAttention_81793357185521
// MI455X (gfx1250) — matrix operand order repaired and checked by reading, not compiled, not run on hardware
//
#include <hip/hip_runtime.h>
#include <hip/hip_fp16.h>
#include <math.h>

// Problem constants (match reference)
#define B_   4
#define T_   2048
#define C_   1024
#define H_   16
#define D_   64
#define N3_  3072   // 3*C

typedef _Float16 half_t;
typedef __attribute__((ext_vector_type(16))) _Float16 v16h;
typedef __attribute__((ext_vector_type(8)))  float    v8f;

// ---- CDNA5 async global->LDS copies (ASYNCcnt-tracked), feature-guarded ----
#if defined(__has_builtin)
#if __has_builtin(__builtin_amdgcn_global_load_async_to_lds_b128)
#define HAVE_ASYNC_LDS 1
#endif
#endif

#if defined(__has_builtin)
#if __has_builtin(__builtin_amdgcn_s_wait_asynccnt)
#define WAIT_ASYNC() __builtin_amdgcn_s_wait_asynccnt(0)
#else
#define WAIT_ASYNC() asm volatile("s_wait_asynccnt 0" ::: "memory")
#endif
#else
#define WAIT_ASYNC() asm volatile("s_wait_asynccnt 0" ::: "memory")
#endif

#if HAVE_ASYNC_LDS
typedef int b128_t __attribute__((vector_size(16)));   // v4i32, matches builtin param
__device__ __forceinline__ void async_copy_b128(const half_t* g, half_t* l) {
    __builtin_amdgcn_global_load_async_to_lds_b128(
        (__attribute__((address_space(1))) b128_t*)g,
        (__attribute__((address_space(3))) b128_t*)l,
        0, 0);
}
#endif

union Frag {            // 8 VGPRs = 16 f16 = two 16B pieces
    v16h   h;
    float4 f4[2];
};
union VPack {           // for f16 extraction from a 16B global load
    float4 f4;
    half_t h[8];
};

__device__ __forceinline__ v8f wmma_f16(v16h a, v16h b, v8f c) {
    // D = A(16x32 f16) * B(32x16 f16) + C(16x16 f32)
    return __builtin_amdgcn_wmma_f32_16x16x32_f16(
        /*neg_a=*/false, a, /*neg_b=*/false, b,
        /*c_mod=*/(short)0, c, /*reuse_a=*/false, /*reuse_b=*/false);
}

// ---- DPP butterfly reductions within each 16-lane row (pure VALU, no LDS) ----
template <int CTRL>
__device__ __forceinline__ float dppmov(float x) {
    return __builtin_bit_cast(float,
        __builtin_amdgcn_update_dpp(0, __builtin_bit_cast(int, x),
                                    CTRL, 0xF, 0xF, true));
}
__device__ __forceinline__ float redmax16(float v) {
    v = fmaxf(v, dppmov<0xB1>(v));   // quad_perm [1,0,3,2]  : xor 1
    v = fmaxf(v, dppmov<0x4E>(v));   // quad_perm [2,3,0,1]  : xor 2
    v = fmaxf(v, dppmov<0x141>(v));  // ROW_HALF_MIRROR      : xor within 8
    v = fmaxf(v, dppmov<0x140>(v));  // ROW_MIRROR           : xor within 16
    return v;
}
__device__ __forceinline__ float redsum16(float v) {
    v += dppmov<0xB1>(v);
    v += dppmov<0x4E>(v);
    v += dppmov<0x141>(v);
    v += dppmov<0x140>(v);
    return v;
}

// ---------------------------------------------------------------------------
// 1) fp32 -> f16 conversion (grid-stride)
// ---------------------------------------------------------------------------
__global__ void cvt_f32_to_f16(const float* __restrict__ src,
                               half_t* __restrict__ dst, int n) {
    int i = blockIdx.x * blockDim.x + threadIdx.x;
    int stride = gridDim.x * blockDim.x;
    for (; i < n; i += stride) dst[i] = (half_t)src[i];
}

// ---------------------------------------------------------------------------
// 2) Generic f16 WMMA GEMM:  Out[M][N] = A[M][K] * B[N][K]^T + bias[N]
//    Block tile 128x128, BK=32, 256 threads = 8 waves (2x4 wave grid),
//    each wave: 64x32 = 4x2 accumulators of 16x16.
//    Staging: async global->LDS with double buffering (CDNA5 path), or
//    register-pipelined loads as fallback.
// ---------------------------------------------------------------------------
template <bool HALF_OUT>
__global__ __launch_bounds__(256)
void gemm_f16_wmma(const half_t* __restrict__ A,
                   const half_t* __restrict__ Bm,
                   const float*  __restrict__ bias,
                   void* __restrict__ Out,
                   int M, int N, int K)
{
    const int LDA = 40;                     // 32 + 8 pad halves (80B, 16B aligned)

    const int tid   = threadIdx.x;
    const int lane  = tid & 31;
    const int wave  = tid >> 5;
    const int wm    = wave & 1;             // 0..1  (64-row groups)
    const int wn    = wave >> 1;            // 0..3  (32-col groups)
    const int hlf   = lane >> 4;            // 0..1
    const int l16   = lane & 15;

    const long m0 = (long)blockIdx.y * 128;
    const long n0 = (long)blockIdx.x * 128;

    v8f acc[4][2];
#pragma unroll
    for (int i = 0; i < 4; ++i)
#pragma unroll
        for (int j = 0; j < 2; ++j)
            acc[i][j] = (v8f){};

#if HAVE_ASYNC_LDS
    // ---------------- async double-buffered staging ----------------
    __shared__ half_t Al[2][128 * 40];
    __shared__ half_t Bl[2][128 * 40];

    // 512 slots of 8 halves (16B) per matrix; thread owns slots tid, tid+256
    auto stage = [&](int buf, int k0) {
#pragma unroll
        for (int i = 0; i < 2; ++i) {
            const int s = tid + 256 * i;
            const int row = s >> 2, seg = s & 3;
            async_copy_b128(A  + (m0 + row) * (long)K + k0 + seg * 8,
                            &Al[buf][row * LDA + seg * 8]);
            async_copy_b128(Bm + (n0 + row) * (long)K + k0 + seg * 8,
                            &Bl[buf][row * LDA + seg * 8]);
        }
    };

    stage(0, 0);
    int buf = 0;
    for (int k0 = 0; k0 < K; k0 += 32, buf ^= 1) {
        WAIT_ASYNC();              // this wave's async fills of `buf` have landed
        __syncthreads();           // all waves done filling `buf` / reading `buf^1`
        if (k0 + 32 < K)
            stage(buf ^ 1, k0 + 32);   // fill other buffer while we compute

        const half_t* Ab = &Al[buf][0];
        const half_t* Bb = &Bl[buf][0];
#else
    // ---------------- register-pipelined staging (fallback) ----------------
    __shared__ half_t Al[128 * 40];
    __shared__ half_t Bl[128 * 40];

    float4 ra[2], rb[2];
#pragma unroll
    for (int i = 0; i < 2; ++i) {
        const int s = tid + 256 * i;
        const int row = s >> 2, seg = s & 3;
        ra[i] = *(const float4*)(A  + (m0 + row) * (long)K + seg * 8);
        rb[i] = *(const float4*)(Bm + (n0 + row) * (long)K + seg * 8);
    }

    for (int k0 = 0; k0 < K; k0 += 32) {
        __syncthreads();
#pragma unroll
        for (int i = 0; i < 2; ++i) {
            const int s = tid + 256 * i;
            const int row = s >> 2, seg = s & 3;
            *(float4*)&Al[row * LDA + seg * 8] = ra[i];
            *(float4*)&Bl[row * LDA + seg * 8] = rb[i];
        }
        __syncthreads();

        if (k0 + 32 < K) {
#pragma unroll
            for (int i = 0; i < 2; ++i) {
                const int s = tid + 256 * i;
                const int row = s >> 2, seg = s & 3;
                ra[i] = *(const float4*)(A  + (m0 + row) * (long)K + (k0 + 32) + seg * 8);
                rb[i] = *(const float4*)(Bm + (n0 + row) * (long)K + (k0 + 32) + seg * 8);
            }
        }
        const half_t* Ab = &Al[0];
        const half_t* Bb = &Bl[0];
#endif

        v16h afr[4], bfr[2];
#pragma unroll
        for (int mi = 0; mi < 4; ++mi) {
            const int r = wm * 64 + mi * 16 + l16;     // A row (lane = M)
            Frag f;
            f.f4[0] = *(const float4*)&Ab[r * LDA + 8 * hlf];        // K = 8h+0..7
            f.f4[1] = *(const float4*)&Ab[r * LDA + 16 + 8 * hlf];   // K = 16+8h+0..7
            afr[mi] = f.h;
        }
#pragma unroll
        for (int ni = 0; ni < 2; ++ni) {
            const int r = wn * 32 + ni * 16 + l16;     // B row (lane = N)
            Frag f;
            f.f4[0] = *(const float4*)&Bb[r * LDA + 8 * hlf];        // K = 8h+0..7
            f.f4[1] = *(const float4*)&Bb[r * LDA + 16 + 8 * hlf];   // K = 16+8h+0..7
            bfr[ni] = f.h;
        }
#pragma unroll
        for (int mi = 0; mi < 4; ++mi)
#pragma unroll
            for (int ni = 0; ni < 2; ++ni)
                acc[mi][ni] = wmma_f16(afr[mi], bfr[ni], acc[mi][ni]);
    }

    // epilogue: C layout -> lane holds column N=l16, rows r + 8*hlf
#pragma unroll
    for (int mi = 0; mi < 4; ++mi) {
#pragma unroll
        for (int ni = 0; ni < 2; ++ni) {
            const long col = n0 + wn * 32 + ni * 16 + l16;
            const float bv = bias[col];
#pragma unroll
            for (int r = 0; r < 8; ++r) {
                const long row = m0 + wm * 64 + mi * 16 + r + 8 * hlf;
                const float v = acc[mi][ni][r] + bv;
                if (HALF_OUT)
                    ((half_t*)Out)[row * N + col] = (half_t)v;
                else
                    ((float*)Out)[row * N + col] = v;
            }
        }
    }
}

// ---------------------------------------------------------------------------
// 3) RoPE + scatter qkv[B*T][3C] (f16) -> Q/K/V [B][H][T][D] (f16)
// ---------------------------------------------------------------------------
__global__ __launch_bounds__(256)
void rope_scatter(const half_t* __restrict__ qkv,
                  half_t* __restrict__ Q,
                  half_t* __restrict__ Ko,
                  half_t* __restrict__ V)
{
    const int idx = blockIdx.x * 256 + threadIdx.x;   // 2^22 total
    const int j = idx & 31;
    const int t = (idx >> 5) & (T_ - 1);
    const int h = (idx >> 16) & (H_ - 1);
    const int b = idx >> 20;

    const float fr = __powf(40.0f, (float)j * (1.0f / 31.0f));
    const float th = 6.2831853071795864f * ((float)t * (1.0f / (float)T_)) * fr;
    float sn, cs;
    __sincosf(th, &sn, &cs);

    const long row = (long)(b * T_ + t) * N3_;
    const int  hd  = h * D_ + j;

    const float q1 = (float)qkv[row + hd],          q2 = (float)qkv[row + hd + 32];
    const float k1 = (float)qkv[row + C_ + hd],     k2 = (float)qkv[row + C_ + hd + 32];
    const float v1 = (float)qkv[row + 2 * C_ + hd], v2 = (float)qkv[row + 2 * C_ + hd + 32];

    const long o = ((long)(b * H_ + h) * T_ + t) * D_ + j;
    Q[o]       = (half_t)(q1 * cs - q2 * sn);
    Q[o + 32]  = (half_t)(q1 * sn + q2 * cs);
    Ko[o]      = (half_t)(k1 * cs - k2 * sn);
    Ko[o + 32] = (half_t)(k1 * sn + k2 * cs);
    V[o]       = (half_t)v1;
    V[o + 32]  = (half_t)v2;
}

// ---------------------------------------------------------------------------
// 4) Flash attention per (b,h). 256 threads = 8 waves, each wave owns a
//    16-query tile. Keys staged 64 at a time (two 32-key WMMA sub-steps per
//    staging), K/V prefetched through registers.
// ---------------------------------------------------------------------------
__global__ __launch_bounds__(256)
void flash_attn(const half_t* __restrict__ Q,
                const half_t* __restrict__ Kt,
                const half_t* __restrict__ V,
                half_t* __restrict__ ctx)
{
    __shared__ half_t Kl[64 * 72];          // [key][d],  stride 72 halves
    __shared__ half_t Vt[64 * 72];          // [d][key],  stride 72 halves
    __shared__ half_t Pl[8 * 16 * 40];      // per-wave 16x32 P tile, stride 40

    const int tid  = threadIdx.x;
    const int lane = tid & 31;
    const int wave = tid >> 5;
    const int hlf  = lane >> 4;
    const int l16  = lane & 15;

    const int  bh   = blockIdx.y;           // b*H + h
    const int  b    = bh >> 4;
    const int  h    = bh & 15;
    const long base = (long)bh * T_ * D_;
    const int  q0   = blockIdx.x * 128 + wave * 16;

    // Q A-fragments (d 0..31 and 32..63), loaded once from global
    v16h qa0, qa1;
    {
        const half_t* qrow = Q + base + (long)(q0 + l16) * D_;
        Frag f;
        f.f4[0] = *(const float4*)(qrow + 8 * hlf);
        f.f4[1] = *(const float4*)(qrow + 16 + 8 * hlf);
        qa0 = f.h;
        f.f4[0] = *(const float4*)(qrow + 32 + 8 * hlf);
        f.f4[1] = *(const float4*)(qrow + 48 + 8 * hlf);
        qa1 = f.h;
    }

    float mrow[8], lrow[8];
    v8f acc[4];
#pragma unroll
    for (int r = 0; r < 8; ++r) { mrow[r] = -1e30f; lrow[r] = 0.0f; }
#pragma unroll
    for (int d = 0; d < 4; ++d) acc[d] = (v8f){};

    const float scale = 0.125f;             // 1/sqrt(64)

    // preload first 64-key slab into registers
    float4 rk[2]; VPack rv[2];
#pragma unroll
    for (int i = 0; i < 2; ++i) {
        const int s = tid + 256 * i;
        const int row = s >> 3, seg = s & 7;
        rk[i]    = *(const float4*)(Kt + base + (long)row * D_ + seg * 8);
        rv[i].f4 = *(const float4*)(V  + base + (long)row * D_ + seg * 8);
    }

    for (int kb = 0; kb < T_; kb += 64) {
        __syncthreads();
#pragma unroll
        for (int i = 0; i < 2; ++i) {
            const int s = tid + 256 * i;
            const int row = s >> 3, seg = s & 7;
            *(float4*)&Kl[row * 72 + seg * 8] = rk[i];     // [key][d]
#pragma unroll
            for (int e = 0; e < 8; ++e)                    // transpose V: [d][key]
                Vt[(seg * 8 + e) * 72 + row] = rv[i].h[e];
        }
        __syncthreads();

        // prefetch next slab while we compute
        if (kb + 64 < T_) {
#pragma unroll
            for (int i = 0; i < 2; ++i) {
                const int s = tid + 256 * i;
                const int row = s >> 3, seg = s & 7;
                rk[i]    = *(const float4*)(Kt + base + (long)(kb + 64 + row) * D_ + seg * 8);
                rv[i].f4 = *(const float4*)(V  + base + (long)(kb + 64 + row) * D_ + seg * 8);
            }
        }

#pragma unroll
        for (int sub = 0; sub < 64; sub += 32) {
            // S tiles: s01[c] covers keys sub+c*16 .. +15, contraction d=64 (2 steps)
            v8f s01[2];
#pragma unroll
            for (int c = 0; c < 2; ++c) {
                const int key = sub + c * 16 + l16;
                Frag f;
                f.f4[0] = *(const float4*)&Kl[key * 72 + 8 * hlf];
                f.f4[1] = *(const float4*)&Kl[key * 72 + 16 + 8 * hlf];
                const v16h b0 = f.h;
                f.f4[0] = *(const float4*)&Kl[key * 72 + 32 + 8 * hlf];
                f.f4[1] = *(const float4*)&Kl[key * 72 + 48 + 8 * hlf];
                const v16h b1 = f.h;
                v8f z = (v8f){};
                z      = wmma_f16(qa0, b0, z);
                s01[c] = wmma_f16(qa1, b1, z);
            }

            // online softmax; row M = r + 8*hlf, columns spread over 16 lanes
            float pr0[8], pr1[8];
#pragma unroll
            for (int r = 0; r < 8; ++r) {
                const float tmax = redmax16(fmaxf(s01[0][r], s01[1][r]) * scale);
                const float mn    = fmaxf(mrow[r], tmax);
                const float alpha = __expf(mrow[r] - mn);
                mrow[r] = mn;
                const float p0 = __expf(s01[0][r] * scale - mn);
                const float p1 = __expf(s01[1][r] * scale - mn);
                const float ps = redsum16(p0 + p1);
                lrow[r] = lrow[r] * alpha + ps;
#pragma unroll
                for (int di = 0; di < 4; ++di) acc[di][r] *= alpha;
                pr0[r] = p0; pr1[r] = p1;
            }

            // P: C-layout -> LDS -> A-fragment layout (same-wave LDS is in-order;
            // fences only pin compiler ordering)
            half_t* pw = &Pl[wave * 16 * 40];
#pragma unroll
            for (int r = 0; r < 8; ++r) {
                const int rr = r + 8 * hlf;
                pw[rr * 40 + l16]      = (half_t)pr0[r];
                pw[rr * 40 + 16 + l16] = (half_t)pr1[r];
            }
            __builtin_amdgcn_fence(__ATOMIC_ACQ_REL, "wavefront");
            v16h pa;
            {
                Frag f;
                f.f4[0] = *(const float4*)&pw[l16 * 40 + 8 * hlf];
                f.f4[1] = *(const float4*)&pw[l16 * 40 + 16 + 8 * hlf];
                pa = f.h;
            }
            __builtin_amdgcn_fence(__ATOMIC_ACQ_REL, "wavefront");

            // O += P(16x32) * V(32x16) for each 16-wide d tile
#pragma unroll
            for (int di = 0; di < 4; ++di) {
                const int d = di * 16 + l16;
                Frag f;
                f.f4[0] = *(const float4*)&Vt[d * 72 + sub + 8 * hlf];
                f.f4[1] = *(const float4*)&Vt[d * 72 + sub + 16 + 8 * hlf];
                acc[di] = wmma_f16(pa, f.h, acc[di]);
            }
        }
    }

    // normalize and write ctx[b*T + q][h*64 + d]
#pragma unroll
    for (int r = 0; r < 8; ++r) {
        const float inv = 1.0f / lrow[r];
        const int   q   = q0 + r + 8 * hlf;
        const long  orow = (long)(b * T_ + q) * C_ + h * D_;
#pragma unroll
        for (int di = 0; di < 4; ++di)
            ctx[orow + di * 16 + l16] = (half_t)(acc[di][r] * inv);
    }
}

// ---------------------------------------------------------------------------
// launch
// ---------------------------------------------------------------------------
extern "C" void kernel_launch(void* const* d_in, const int* in_sizes, int n_in,
                              void* d_out, int out_size, void* d_ws, size_t ws_size,
                              hipStream_t stream)
{
    (void)in_sizes; (void)n_in; (void)out_size; (void)ws_size;

    const float* x      = (const float*)d_in[0];   // [B,T,C]
    const float* qkv_w  = (const float*)d_in[1];   // [3C,C]
    const float* qkv_b  = (const float*)d_in[2];   // [3C]
    const float* proj_w = (const float*)d_in[3];   // [C,C]
    const float* proj_b = (const float*)d_in[4];   // [C]
    float*       out    = (float*)d_out;           // [B,T,C]

    // workspace layout (f16), total ~136 MB
    const long nX   = (long)B_ * T_ * C_;          //  8388608
    const long nWq  = (long)N3_ * C_;              //  3145728
    const long nWp  = (long)C_ * C_;               //  1048576
    const long nQKV = (long)B_ * T_ * N3_;         // 25165824
    const long nH   = (long)B_ * H_ * T_ * D_;     //  8388608

    half_t* ws   = (half_t*)d_ws;
    half_t* xh   = ws;                 ws += nX;
    half_t* wqh  = ws;                 ws += nWq;
    half_t* wph  = ws;                 ws += nWp;
    half_t* qkvh = ws;                 ws += nQKV;
    half_t* Qh   = ws;                 ws += nH;
    half_t* Kh   = ws;                 ws += nH;
    half_t* Vh   = ws;                 ws += nH;
    half_t* ctxh = ws;                 ws += nX;

    // 1) convert inputs to f16
    cvt_f32_to_f16<<<(int)(nX  / 256), 256, 0, stream>>>(x,      xh,  (int)nX);
    cvt_f32_to_f16<<<(int)(nWq / 256), 256, 0, stream>>>(qkv_w,  wqh, (int)nWq);
    cvt_f32_to_f16<<<(int)(nWp / 256), 256, 0, stream>>>(proj_w, wph, (int)nWp);

    // 2) QKV = xh @ wqh^T + qkv_b  -> f16 [8192][3072]
    {
        dim3 grid(N3_ / 128, (B_ * T_) / 128);     // (24, 64)
        gemm_f16_wmma<true><<<grid, 256, 0, stream>>>(
            xh, wqh, qkv_b, (void*)qkvh, B_ * T_, N3_, C_);
    }

    // 3) RoPE + scatter into Q/K/V [B][H][T][D]
    rope_scatter<<<(B_ * H_ * T_ * 32) / 256, 256, 0, stream>>>(qkvh, Qh, Kh, Vh);

    // 4) flash attention -> ctxh [8192][1024]
    {
        dim3 grid(T_ / 128, B_ * H_);              // (16, 64)
        flash_attn<<<grid, 256, 0, stream>>>(Qh, Kh, Vh, ctxh);
    }

    // 5) out = ctxh @ wph^T + proj_b  -> f32 [8192][1024]
    {
        dim3 grid(C_ / 128, (B_ * T_) / 128);      // (8, 64)
        gemm_f16_wmma<false><<<grid, 256, 0, stream>>>(
            ctxh, wph, proj_b, (void*)out, B_ * T_, C_, C_);
    }
}
